// BAtten_17214228922470
// MI455X (gfx1250) — matrix operand order repaired and checked by reading, not compiled, not run on hardware
//
#include <hip/hip_runtime.h>

// ---------------- problem constants ----------------
#define DIMC  256
#define HImg  120
#define WImg  120
#define BImg  4
#define NPOS  (BImg*HImg*WImg)   // 57600
#define NHEAD 8
#define DH    32
#define WINS  5
#define NB1   24
#define NWIN  (BImg*NB1*NB1)     // 2304
#define NKEY  225
#define NQ    25
#define LDP   48                 // padded row length (ushorts) = 96B, 16B-aligned
#define LDF   36                 // padded f32 row length = 144B, 16B-aligned

// ---------------- WMMA / vector types ----------------
typedef __attribute__((ext_vector_type(16))) __bf16        v16bf;
typedef __attribute__((ext_vector_type(8)))  float         v8f;
typedef __attribute__((ext_vector_type(4)))  unsigned int  v4u;
typedef __attribute__((ext_vector_type(4)))  unsigned int  v4ui;
typedef __attribute__((ext_vector_type(8)))  int           v8ii;
typedef __attribute__((ext_vector_type(4)))  int           v4ii;

union FragU { v16bf v; v4u q[2]; unsigned int w[8]; unsigned short u[16]; };

__device__ __forceinline__ unsigned short f2bf(float f) {
  unsigned int u = __float_as_uint(f);
  u += 0x7fffu + ((u >> 16) & 1u);          // round-to-nearest-even
  return (unsigned short)(u >> 16);
}
__device__ __forceinline__ unsigned pack2bf(float lo, float hi) {
  return (unsigned)f2bf(lo) | ((unsigned)f2bf(hi) << 16);
}
__device__ __forceinline__ float bf2f(unsigned short h) {
  return __uint_as_float(((unsigned int)h) << 16);
}

__device__ __forceinline__ v8f wmma_bf16(v16bf a, v16bf b, v8f c) {
  return __builtin_amdgcn_wmma_f32_16x16x32_bf16(false, a, false, b,
                                                 (short)0, c, false, false);
}

// ---- async global->LDS staging (CDNA5 GV mode), tracked by ASYNCcnt ----
__device__ __forceinline__ void async_b128_to_lds(void* lds_ptr, const void* gptr) {
  unsigned lds_off = (unsigned)(size_t)lds_ptr;  // ISA 10.2: LDS addr = addr[31:0]
  asm volatile("global_load_async_to_lds_b128 %0, %1, off"
               :: "v"(lds_off), "v"(gptr) : "memory");
}
__device__ __forceinline__ void wait_asynccnt0() {
  asm volatile("s_wait_asynccnt 0x0" ::: "memory");
}

// A fragment: 16x32 bf16 from row-major [ld ushorts] tile -> 2x ds_load_b128.
__device__ __forceinline__ v16bf load_a_frag(const unsigned short* base, int ld,
                                             int row0, int lane) {
  FragU a;
  const unsigned short* p = base + (row0 + (lane & 15)) * ld + ((lane >> 4) & 1) * 16;
  a.q[0] = *(const v4u*)(p);
  a.q[1] = *(const v4u*)(p + 8);
  return a.v;
}

// B fragment from [N][K] bf16 tile -> 2x ds_load_b128.
__device__ __forceinline__ v16bf load_bT_frag(const unsigned short* base, int ld,
                                              int col0, int lane) {
  FragU b;
  const unsigned short* p = base + (col0 + (lane & 15)) * ld + ((lane >> 4) & 1) * 16;
  b.q[0] = *(const v4u*)(p);
  b.q[1] = *(const v4u*)(p + 8);
  return b.v;
}

// B fragment from a raw-f32 [N][K] tile, converting while filling VGPRs.
__device__ __forceinline__ v16bf load_bT_frag_f32(const float* base, int ld,
                                                  int col0, int lane) {
  FragU b;
  const float* p = base + (col0 + (lane & 15)) * ld + ((lane >> 4) & 1) * 16;
#pragma unroll
  for (int i = 0; i < 8; ++i) b.w[i] = pack2bf(p[2 * i], p[2 * i + 1]);
  return b.v;
}

// =====================================================================
// Kernel 1: xq/xk/xv = x @ W{q,k,v}^T  (bf16 out, [pos][c] row-major)
// =====================================================================
__global__ __launch_bounds__(256) void proj3_kernel(
    const float* __restrict__ x,
    const float* __restrict__ Wq, const float* __restrict__ Wk,
    const float* __restrict__ Wv,
    unsigned short* __restrict__ xq, unsigned short* __restrict__ xk,
    unsigned short* __restrict__ xv) {
  __shared__ __align__(16) unsigned short As[64][LDP];   // [M][K]
  __shared__ __align__(16) unsigned short Bs[64][LDP];   // [N][K] (frag-contiguous)

  const int tid  = threadIdx.x;
  const int wave = tid >> 5, lane = tid & 31;
  const int z    = blockIdx.z;
  const float* Wm = (z == 0) ? Wq : (z == 1 ? Wk : Wv);
  unsigned short* dst = (z == 0) ? xq : (z == 1 ? xk : xv);
  const float scale = (z == 0) ? 0.17677669529663687f : 1.0f;  // dh^-0.5

  const int mBase = blockIdx.x * 64;
  const int nBase = blockIdx.y * 64;
  const int mt = (wave & 3) * 16;
  const int nt = (wave >> 2) * 32;

  v8f acc0 = {}; v8f acc1 = {};

  for (int k0 = 0; k0 < DIMC; k0 += 32) {
    if (k0 + 32 < DIMC)
      __builtin_prefetch(&Wm[(nBase) * DIMC + k0 + 32], 0, 1);
    // stage A: transposed gather from NCHW fp32, packed bf16 pair stores
#pragma unroll
    for (int p = 0; p < 4; ++p) {
      int idx = p * 256 + tid;
      int mr = idx & 63, kp = idx >> 6;           // kp = 0..15 (pairs)
      int m = mBase + mr;
      int bi = m / (HImg * WImg);
      int hw = m - bi * (HImg * WImg);
      const float* xp = x + ((long)(bi * DIMC + k0 + 2 * kp)) * (HImg * WImg) + hw;
      *(unsigned*)&As[mr][2 * kp] = pack2bf(xp[0], xp[HImg * WImg]);
    }
    // stage B: Bs[j][k] = W[j][k] (no transpose; coalesced pair loads)
#pragma unroll
    for (int p = 0; p < 4; ++p) {
      int idx = p * 256 + tid;
      int kp = idx & 15, jc = idx >> 4;
      const float* wp = Wm + (long)(nBase + jc) * DIMC + k0 + 2 * kp;
      *(unsigned*)&Bs[jc][2 * kp] = pack2bf(wp[0], wp[1]);
    }
    __syncthreads();

    v16bf af = load_a_frag(&As[0][0], LDP, mt, lane);
    v16bf b0 = load_bT_frag(&Bs[0][0], LDP, nt, lane);
    v16bf b1 = load_bT_frag(&Bs[0][0], LDP, nt + 16, lane);
    acc0 = wmma_bf16(af, b0, acc0);
    acc1 = wmma_bf16(af, b1, acc1);
    __syncthreads();
  }

  const int hi = (lane >> 4) & 1;
#pragma unroll
  for (int r = 0; r < 8; ++r) {
    int m  = mBase + mt + r + hi * 8;
    int n0 = nBase + nt + (lane & 15);
    dst[(long)m * DIMC + n0]      = f2bf(acc0[r] * scale);
    dst[(long)m * DIMC + n0 + 16] = f2bf(acc1[r] * scale);
  }
}

// =====================================================================
// Kernel 2: windowed attention. One block per window, one wave per head.
// Conv templates applied post-projection (commutes with channel mixing).
// Kb doubles as the P buffer (K dead after S-WMMA, same-wave DS order).
// =====================================================================
__global__ __launch_bounds__(256) void atten_kernel(
    const unsigned short* __restrict__ xq,
    const unsigned short* __restrict__ xk,
    const unsigned short* __restrict__ xv,
    const float* __restrict__ conv_w,
    unsigned short* __restrict__ ao) {
  __shared__ float twt[9][9];
  __shared__ __align__(16) unsigned short Kb[NHEAD][32][LDP];   // K: [key][ch]; later P: [q][key]
  __shared__ __align__(16) unsigned short Vbt[NHEAD][32][LDP];  // V transposed: [ch][key]

  const int tid = threadIdx.x, wave = tid >> 5, lane = tid & 31;
  if (tid < 81) twt[tid / 9][tid % 9] = conv_w[(tid / 9) * (DIMC * 9) + (tid % 9)];
  __syncthreads();

  const int wi = blockIdx.x;
  const int bi = wi / (NB1 * NB1);
  const int rw = wi - bi * (NB1 * NB1);
  const int b1 = rw / NB1, b2 = rw - b1 * NB1;
  const int h0 = b1 * WINS, w0 = b2 * WINS;
  const int cb = wave * DH;
  const int hi = (lane >> 4) & 1;

  // ---- Q A-fragments, straight from global (2x b128 per tile) ----
  v16bf qf[2];
#pragma unroll
  for (int mt2 = 0; mt2 < 2; ++mt2) {
    FragU a;
    int row = mt2 * 16 + (lane & 15);
    if (row < NQ) {
      int ph = row / WINS, pw = row - ph * WINS;
      long pos = (long)bi * (HImg * WImg) + (h0 + ph) * WImg + (w0 + pw);
      const unsigned short* qp = xq + pos * DIMC + cb + hi * 16;
      a.q[0] = *(const v4u*)(qp);
      a.q[1] = *(const v4u*)(qp + 8);
    } else {
      v4u zq = {};
      a.q[0] = zq; a.q[1] = zq;
    }
    qf[mt2] = a.v;
  }

  v8f of[2][2];
  v8f zf = {};
  of[0][0] = zf; of[0][1] = zf; of[1][0] = zf; of[1][1] = zf;
  float mrun[2][8], lrun[2][8];
#pragma unroll
  for (int i = 0; i < 2; ++i)
#pragma unroll
    for (int r = 0; r < 8; ++r) { mrun[i][r] = -1e30f; lrun[i][r] = 0.0f; }

  for (int kb = 0; kb < 8; ++kb) {
    // ---- build 32-key K/V block via on-the-fly 3x3 conv (lane = channel) ----
    for (int rloc = 0; rloc < 32; ++rloc) {
      int kk = kb * 32 + rloc;
      float ak = 0.0f, av = 0.0f;
      if (kk < NKEY) {
        int t = kk / NQ, p = kk - t * NQ;
        int ph = h0 + p / WINS, pw = w0 + (p - (p / WINS) * WINS);
#pragma unroll
        for (int dy = 0; dy < 3; ++dy) {
          int hh = ph + dy - 1;
#pragma unroll
          for (int dx = 0; dx < 3; ++dx) {
            int ww = pw + dx - 1;
            if (hh >= 0 && hh < HImg && ww >= 0 && ww < WImg) {
              long pos = (long)bi * (HImg * WImg) + hh * WImg + ww;
              float wgt = twt[t][dy * 3 + dx];
              ak += wgt * bf2f(xk[pos * DIMC + cb + lane]);
              av += wgt * bf2f(xv[pos * DIMC + cb + lane]);
            }
          }
        }
      }
      Kb[wave][rloc][lane]  = f2bf(ak);   // [key][ch]
      Vbt[wave][lane][rloc] = f2bf(av);   // [ch][key]
    }

    // ---- S = Q*K^T : B[k=ch][n=key], Kb is [key][ch] == [N][K] ----
    v8f sf[2][2];
#pragma unroll
    for (int mt2 = 0; mt2 < 2; ++mt2) {
#pragma unroll
      for (int ntile = 0; ntile < 2; ++ntile) {
        v16bf bfr = load_bT_frag(&Kb[wave][0][0], LDP, ntile * 16, lane);
        v8f zacc = {};
        sf[mt2][ntile] = wmma_bf16(qf[mt2], bfr, zacc);
      }
    }

    // ---- mask padded keys ----
#pragma unroll
    for (int ntile = 0; ntile < 2; ++ntile) {
      int key = kb * 32 + ntile * 16 + (lane & 15);
      if (key >= NKEY) {
#pragma unroll
        for (int r = 0; r < 8; ++r) { sf[0][ntile][r] = -1e30f; sf[1][ntile][r] = -1e30f; }
      }
    }

    // ---- online softmax (row reductions within 16-lane halves) ----
#pragma unroll
    for (int mt2 = 0; mt2 < 2; ++mt2) {
#pragma unroll
      for (int r = 0; r < 8; ++r) {
        float v = fmaxf(sf[mt2][0][r], sf[mt2][1][r]);
#pragma unroll
        for (int off = 1; off < 16; off <<= 1) v = fmaxf(v, __shfl_xor(v, off, 32));
        float mnew  = fmaxf(mrun[mt2][r], v);
        float alpha = __expf(mrun[mt2][r] - mnew);
        mrun[mt2][r] = mnew;
        float p0 = __expf(sf[mt2][0][r] - mnew);
        float p1 = __expf(sf[mt2][1][r] - mnew);
        float s = p0 + p1;
#pragma unroll
        for (int off = 1; off < 16; off <<= 1) s += __shfl_xor(s, off, 32);
        lrun[mt2][r] = lrun[mt2][r] * alpha + s;
        of[mt2][0][r] *= alpha;
        of[mt2][1][r] *= alpha;
        int prow = mt2 * 16 + r + hi * 8;     // C-frag lane mapping
        Kb[wave][prow][(lane & 15)]      = f2bf(p0);   // P overwrites K
        Kb[wave][prow][16 + (lane & 15)] = f2bf(p1);
      }
    }

    // ---- O += P * V : A = P [q][key] (in Kb), B from Vbt [ch][key] ----
#pragma unroll
    for (int mt2 = 0; mt2 < 2; ++mt2) {
      v16bf pa = load_a_frag(&Kb[wave][0][0], LDP, mt2 * 16, lane);
#pragma unroll
      for (int ct = 0; ct < 2; ++ct) {
        v16bf vb = load_bT_frag(&Vbt[wave][0][0], LDP, ct * 16, lane);
        of[mt2][ct] = wmma_bf16(pa, vb, of[mt2][ct]);
      }
    }
  }

  // ---- finalize: divide by l, store [window][query][c] bf16 ----
#pragma unroll
  for (int mt2 = 0; mt2 < 2; ++mt2) {
#pragma unroll
    for (int r = 0; r < 8; ++r) {
      int row = mt2 * 16 + r + hi * 8;
      float inv = 1.0f / lrun[mt2][r];
      if (row < NQ) {
        long obase = ((long)wi * NQ + row) * DIMC + cb;
        ao[obase + (lane & 15)]      = f2bf(of[mt2][0][r] * inv);
        ao[obase + 16 + (lane & 15)] = f2bf(of[mt2][1][r] * inv);
      }
    }
  }
}

// =====================================================================
// Kernel 3: out = ao @ Wout^T + bout, window-merged scatter into NCHW f32.
// A-tile: Tensor Data Mover descriptor (pad_enable inserts the 32B row pad),
// fallback = GLOBAL_LOAD_ASYNC_TO_LDS_B128. B-tile: async raw-f32 staging,
// converted to bf16 during fragment construction.
// =====================================================================
__global__ __launch_bounds__(256) void outproj_kernel(
    const unsigned short* __restrict__ ao,
    const float* __restrict__ Wout,
    const float* __restrict__ bout,
    float* __restrict__ out) {
  __shared__ __align__(16) unsigned short As[64][LDP];   // [M][K] bf16
  __shared__ __align__(16) float          Braw[64][LDF]; // [N][K] raw f32

  const int tid = threadIdx.x, wave = tid >> 5, lane = tid & 31;
  const int mBase = blockIdx.x * 64, nBase = blockIdx.y * 64;
  const int mt = (wave & 3) * 16, nt = (wave >> 2) * 32;

  v8f acc0 = {}; v8f acc1 = {};

  for (int k0 = 0; k0 < DIMC; k0 += 32) {
    // ---- stage A tile (64 x 32 bf16, row stride DIMC) ----
#if __has_builtin(__builtin_amdgcn_tensor_load_to_lds)
    if (wave == 0) {
      unsigned long ga = (unsigned long)(ao + (size_t)mBase * DIMC + k0);
      unsigned lds = (unsigned)(size_t)&As[0][0];
      v4ui g0;
      g0[0] = 1u;                                    // count=1 (valid user D#)
      g0[1] = lds;                                   // lds_addr
      g0[2] = (unsigned)ga;                          // global_addr[31:0]
      g0[3] = (unsigned)((ga >> 32) & 0x01FFFFFFu) | (2u << 30);  // addr[56:32] | type=2
      v8ii g1;
      g1[0] = (int)((1u << 16) | (1u << 20) | (3u << 22) | (7u << 25));
      //        data_size=2B | pad_enable | pad_interval=16DW(64B) | pad_amount=8DW(32B)
      g1[1] = (int)(((unsigned)DIMC & 0xFFFFu) << 16);            // tensor_dim0 lo16
      g1[2] = (int)(((unsigned)DIMC >> 16) |
                    (((unsigned)NPOS & 0xFFFFu) << 16));          // dim0 hi | dim1 lo
      g1[3] = (int)(((unsigned)NPOS >> 16) | (32u << 16));        // dim1 hi | tile_dim0=32
      g1[4] = 64;                                                 // tile_dim1=64, tile_dim2=0
      g1[5] = DIMC;                                               // tensor_dim0_stride lo32
      g1[6] = 0;
      g1[7] = 0;
      v4ii gz4 = {0, 0, 0, 0};
      v8ii gz8 = {0, 0, 0, 0, 0, 0, 0, 0};
      __builtin_amdgcn_tensor_load_to_lds(g0, g1, gz4, gz4, gz8, 0);
    }
#else
    {
      int row = tid >> 2, seg = tid & 3;  // 4 x 16B segments per row
      const unsigned short* gsrc =
          ao + (size_t)(mBase + row) * DIMC + k0 + seg * 8;
      async_b128_to_lds(&As[row][seg * 8], gsrc);
    }
#endif
    // ---- stage B tile raw f32 via async copy (64 rows x 128B) ----
    {
      int row = tid >> 2, seg = tid & 3;
      const float* wsrc = Wout + (size_t)(nBase + row) * DIMC + k0;
      async_b128_to_lds(&Braw[row][seg * 4], wsrc + seg * 4);
      async_b128_to_lds(&Braw[row][(seg + 4) * 4], wsrc + (seg + 4) * 4);
    }
    wait_asynccnt0();                       // my async writes landed
#if __has_builtin(__builtin_amdgcn_tensor_load_to_lds)
    if (wave == 0) __builtin_amdgcn_s_wait_tensorcnt(0);
#endif
    __syncthreads();                        // everyone's writes visible

    v16bf af = load_a_frag(&As[0][0], LDP, mt, lane);
    v16bf b0 = load_bT_frag_f32(&Braw[0][0], LDF, nt, lane);
    v16bf b1 = load_bT_frag_f32(&Braw[0][0], LDF, nt + 16, lane);
    acc0 = wmma_bf16(af, b0, acc0);
    acc1 = wmma_bf16(af, b1, acc1);
    __syncthreads();
  }

  const int hi = (lane >> 4) & 1;
#pragma unroll
  for (int r = 0; r < 8; ++r) {
    int m  = mBase + mt + r + hi * 8;
    int wi = m / NQ, p = m - wi * NQ;
    int bi = wi / (NB1 * NB1);
    int rw = wi - bi * (NB1 * NB1);
    int b1i = rw / NB1, b2i = rw - b1i * NB1;
    int hh = b1i * WINS + p / WINS;
    int ww = b2i * WINS + (p - (p / WINS) * WINS);
#pragma unroll
    for (int half = 0; half < 2; ++half) {
      int n = nBase + nt + half * 16 + (lane & 15);
      float v = (half ? acc1[r] : acc0[r]) + bout[n];
      out[(((long)bi * DIMC + n) * HImg + hh) * WImg + ww] = v;
    }
  }
}

// =====================================================================
extern "C" void kernel_launch(void* const* d_in, const int* in_sizes, int n_in,
                              void* d_out, int out_size, void* d_ws, size_t ws_size,
                              hipStream_t stream) {
  (void)in_sizes; (void)n_in; (void)out_size; (void)ws_size;
  const float* x      = (const float*)d_in[0];
  const float* conv_w = (const float*)d_in[1];
  const float* Wk     = (const float*)d_in[2];
  const float* Wv     = (const float*)d_in[3];
  const float* Wq     = (const float*)d_in[4];
  const float* Wout   = (const float*)d_in[5];
  const float* bout   = (const float*)d_in[6];
  float* out = (float*)d_out;

  unsigned short* xq = (unsigned short*)d_ws;
  unsigned short* xk = xq + (size_t)NPOS * DIMC;
  unsigned short* xv = xk + (size_t)NPOS * DIMC;
  unsigned short* ao = xv + (size_t)NPOS * DIMC;

  proj3_kernel<<<dim3(NPOS / 64, DIMC / 64, 3), 256, 0, stream>>>(
      x, Wq, Wk, Wv, xq, xk, xv);
  atten_kernel<<<NWIN, 256, 0, stream>>>(xq, xk, xv, conv_w, ao);
  outproj_kernel<<<dim3(NPOS / 64, DIMC / 64), 256, 0, stream>>>(
      ao, Wout, bout, out);
}
